// DMSD_10239202034262
// MI455X (gfx1250) — compile-verified, matrix operand order checked by reading, not run on hardware
//
#include <hip/hip_runtime.h>
#include <cstddef>

// ---------------- types ----------------
typedef __attribute__((ext_vector_type(8)))  float   v8f;
typedef __attribute__((ext_vector_type(16))) __bf16  v16bf;
typedef __attribute__((ext_vector_type(8)))  __bf16  v8bf;

#define HDIM   768
#define FFDIM  3072
#define NHEADS 12
#define SEQ1   256
#define SEQ2   197
#define BATCH  64
#define M1     (BATCH * SEQ1)   // 16384
#define M2     (BATCH * SEQ2)   // 12608

#define TILE_M 128
#define TILE_N 128
#define TILE_K 32
#define LDSTR  40   // 32 + 8 halfs pad (80 bytes/row, 16B aligned)

// ---- CDNA5 async global->LDS copy (ASYNCcnt-tracked), 16 bytes per lane ----
__device__ __forceinline__ unsigned lds_off(const void* p) {
  // generic -> LDS addrspace(3) -> 32-bit LDS byte offset
  return (unsigned)(unsigned long long)(__attribute__((address_space(3))) const void*)p;
}
__device__ __forceinline__ void async_cp16(unsigned lds_byte_off, const void* gaddr) {
  asm volatile("global_load_async_to_lds_b128 %0, %1, off"
               :: "v"(lds_byte_off), "v"(gaddr) : "memory");
}
__device__ __forceinline__ void wait_async0() {
  asm volatile("s_wait_asynccnt 0" ::: "memory");
}

// ---------------- elementwise: f32 -> bf16 ----------------
__global__ void k_f2bf(const float* __restrict__ in, __bf16* __restrict__ out, int n) {
  int i = blockIdx.x * 256 + threadIdx.x;
  if (i < n) out[i] = (__bf16)in[i];
}

// ---------------- weight convert: fp32 [K,N] -> bf16 [N,K] (transposed) ----------------
__global__ void k_wt_convert(const float* __restrict__ in, __bf16* __restrict__ out,
                             int K, int N) {
  int i = blockIdx.x * 256 + threadIdx.x;
  if (i < K * N) {
    int n = i / K;
    int k = i - n * K;
    out[i] = (__bf16)in[(size_t)k * N + n];
  }
}

// ---------------- WMMA bf16 GEMM: C[M,N] = A[M,K] @ Wt[N,K]^T + bias ----------------
// A row-major [M,K] bf16; Wt row-major [N,K] bf16 (i.e. W transposed).
// Double-buffered LDS, async global->LDS staging, 8 WMMAs / wave / k-step.
// Epilogue: +bias, optional exact-erf gelu; writes fp32 (Cf) and/or bf16 (Cbf).
__global__ __launch_bounds__(256) void k_gemm_wmma(
    const __bf16* __restrict__ A, const __bf16* __restrict__ Wt,
    const float* __restrict__ bias, float* __restrict__ Cf,
    __bf16* __restrict__ Cbf, int M, int N, int K, int act_gelu)
{
  __shared__ __bf16 ldsA[2][TILE_M * LDSTR];
  __shared__ __bf16 ldsB[2][TILE_N * LDSTR];

  const int tid  = threadIdx.x;
  const int lane = tid & 31;
  const int wave = tid >> 5;
  const int wm   = wave & 1;    // 64-row group
  const int wn   = wave >> 1;   // 32-col group
  const int m0   = blockIdx.y * TILE_M;
  const int n0   = blockIdx.x * TILE_N;

  v8f acc[4][2];
  v8f zf;
#pragma unroll
  for (int i = 0; i < 8; ++i) zf[i] = 0.0f;
#pragma unroll
  for (int t = 0; t < 4; ++t)
#pragma unroll
    for (int u = 0; u < 2; ++u) acc[t][u] = zf;

  // staging assignment: thread -> (tile row, 16-half chunk)
  const int lr = tid >> 1;          // 0..127
  const int lc = (tid & 1) * 16;    // 0 or 16
  // A rows clamped: OOB rows only feed masked-out C rows, so duplicate data is safe.
  const int grA = (m0 + lr < M) ? (m0 + lr) : (M - 1);
  const int grB = n0 + lr;          // N is always a multiple of TILE_N here
  const __bf16* gA = A  + (size_t)grA * K + lc;
  const __bf16* gB = Wt + (size_t)grB * K + lc;

  const unsigned la0 = lds_off(&ldsA[0][lr * LDSTR + lc]);
  const unsigned la1 = lds_off(&ldsA[1][lr * LDSTR + lc]);
  const unsigned lb0 = lds_off(&ldsB[0][lr * LDSTR + lc]);
  const unsigned lb1 = lds_off(&ldsB[1][lr * LDSTR + lc]);

  const int nk = K / TILE_K;

  // prologue: stage k-step 0 into buffer 0
  async_cp16(la0,      gA);
  async_cp16(la0 + 16, gA + 8);
  async_cp16(lb0,      gB);
  async_cp16(lb0 + 16, gB + 8);

  const int kb   = (lane >> 4) * 8; // per-lane K octet base: 0 or 8
  const int mrow = lane & 15;

  for (int kk = 0; kk < nk; ++kk) {
    wait_async0();        // this wave's copies into buf[kk&1] have landed
    __syncthreads();      // everyone's copies landed; everyone done reading buf[(kk+1)&1]

    if (kk + 1 < nk) {    // stage next k-step into the other buffer (overlaps compute)
      const __bf16* nA = gA + (size_t)(kk + 1) * TILE_K;
      const __bf16* nB = gB + (size_t)(kk + 1) * TILE_K;
      unsigned la = ((kk + 1) & 1) ? la1 : la0;
      unsigned lb = ((kk + 1) & 1) ? lb1 : lb0;
      async_cp16(la,      nA);
      async_cp16(la + 16, nA + 8);
      async_cp16(lb,      nB);
      async_cp16(lb + 16, nB + 8);
    }

    const __bf16* baseA = &ldsA[kk & 1][0];
    const __bf16* baseB = &ldsB[kk & 1][0];

    v16bf afr[4], bfr[2];
#pragma unroll
    for (int t = 0; t < 4; ++t) {
      const __bf16* p = baseA + (wm * 64 + t * 16 + mrow) * LDSTR + kb;
      v8bf lo = *(const v8bf*)p;          // K = kb..kb+7
      v8bf hi = *(const v8bf*)(p + 16);   // K = kb+16..kb+23
      afr[t] = __builtin_shufflevector(lo, hi, 0,1,2,3,4,5,6,7,8,9,10,11,12,13,14,15);
    }
#pragma unroll
    for (int u = 0; u < 2; ++u) {
      const __bf16* p = baseB + (wn * 32 + u * 16 + mrow) * LDSTR + kb;
      v8bf lo = *(const v8bf*)p;
      v8bf hi = *(const v8bf*)(p + 16);
      bfr[u] = __builtin_shufflevector(lo, hi, 0,1,2,3,4,5,6,7,8,9,10,11,12,13,14,15);
    }
#pragma unroll
    for (int t = 0; t < 4; ++t)
#pragma unroll
      for (int u = 0; u < 2; ++u)
        acc[t][u] = __builtin_amdgcn_wmma_f32_16x16x32_bf16(
            false, afr[t], false, bfr[u], (short)0, acc[t][u], false, false);
  }

  // ---- epilogue: C layout M = e + 8*(lane>=16), N = lane%16 ----
  const int lhalf = lane >> 4;
  const int ncol  = lane & 15;
#pragma unroll
  for (int t = 0; t < 4; ++t) {
#pragma unroll
    for (int u = 0; u < 2; ++u) {
      int col = n0 + wn * 32 + u * 16 + ncol;
      float bv = bias ? bias[col] : 0.0f;
#pragma unroll
      for (int e = 0; e < 8; ++e) {
        int row = m0 + wm * 64 + t * 16 + e + lhalf * 8;
        if (row < M) {
          float val = acc[t][u][e] + bv;
          if (act_gelu)
            val = 0.5f * val * (1.0f + erff(val * 0.70710678118654752f));
          if (Cf)  Cf[(size_t)row * N + col]  = val;
          if (Cbf) Cbf[(size_t)row * N + col] = (__bf16)val;
        }
      }
    }
  }
}

// ---------------- residual + LayerNorm over H=768, f32 + bf16 out ----------------
__global__ __launch_bounds__(256) void k_ln_residual(
    const float* __restrict__ x, const float* __restrict__ res,
    const float* __restrict__ w, const float* __restrict__ b,
    float* __restrict__ outf, __bf16* __restrict__ outbf)
{
  __shared__ float red[256];
  const int row = blockIdx.x;
  const int tid = threadIdx.x;
  const float* xr = x   + (size_t)row * HDIM;
  const float* rr = res + (size_t)row * HDIM;

  float v0 = xr[tid]       + rr[tid];
  float v1 = xr[tid + 256] + rr[tid + 256];
  float v2 = xr[tid + 512] + rr[tid + 512];

  red[tid] = v0 + v1 + v2;
  __syncthreads();
  for (int off = 128; off > 0; off >>= 1) {
    if (tid < off) red[tid] += red[tid + off];
    __syncthreads();
  }
  float mean = red[0] * (1.0f / HDIM);
  __syncthreads();

  float d0 = v0 - mean, d1 = v1 - mean, d2 = v2 - mean;
  red[tid] = d0 * d0 + d1 * d1 + d2 * d2;
  __syncthreads();
  for (int off = 128; off > 0; off >>= 1) {
    if (tid < off) red[tid] += red[tid + off];
    __syncthreads();
  }
  float rstd = rsqrtf(red[0] * (1.0f / HDIM) + 1e-12f);

  float y0 = w[tid]       * (d0 * rstd) + b[tid];
  float y1 = w[tid + 256] * (d1 * rstd) + b[tid + 256];
  float y2 = w[tid + 512] * (d2 * rstd) + b[tid + 512];

  float* of = outf + (size_t)row * HDIM;
  of[tid] = y0; of[tid + 256] = y1; of[tid + 512] = y2;
  if (outbf) {
    __bf16* ob = outbf + (size_t)row * HDIM;
    ob[tid] = (__bf16)y0; ob[tid + 256] = (__bf16)y1; ob[tid + 512] = (__bf16)y2;
  }
}

// ---------------- fused flash-style cross attention ----------------
// grid = B*NH blocks, 256 threads; thread = one q row; K/V head in LDS (bf16).
__global__ __launch_bounds__(256) void k_attention(
    const __bf16* __restrict__ q, const __bf16* __restrict__ k,
    const __bf16* __restrict__ v, const float* __restrict__ mask,
    float* __restrict__ ctx)
{
  __shared__ __bf16 Kl[SEQ2 * 64];
  __shared__ __bf16 Vl[SEQ2 * 64];
  const int bh = blockIdx.x;
  const int b  = bh / NHEADS;
  const int h  = bh % NHEADS;
  const int tid = threadIdx.x;

  const size_t kvbase = (size_t)b * SEQ2 * HDIM + h * 64;
  for (int idx = tid; idx < SEQ2 * 64; idx += 256) {
    int l2 = idx >> 6, d = idx & 63;
    Kl[idx] = k[kvbase + (size_t)l2 * HDIM + d];
    Vl[idx] = v[kvbase + (size_t)l2 * HDIM + d];
  }
  __syncthreads();

  const size_t qbase = ((size_t)b * SEQ1 + tid) * HDIM + h * 64;
  float qr[64];
#pragma unroll
  for (int d = 0; d < 64; ++d) qr[d] = (float)q[qbase + d];

  const float* mrow = mask + (size_t)b * SEQ2;
  float m = -1e30f, sum = 0.0f;
  float c[64];
#pragma unroll
  for (int d = 0; d < 64; ++d) c[d] = 0.0f;

  for (int l2 = 0; l2 < SEQ2; ++l2) {
    const __bf16* kp = &Kl[l2 * 64];
    float s = 0.0f;
#pragma unroll
    for (int d = 0; d < 64; ++d) s += qr[d] * (float)kp[d];
    s = s * 0.125f + mrow[l2];            // 1/sqrt(64)
    float mn  = fmaxf(m, s);
    float esc = __expf(m - mn);
    float p   = __expf(s - mn);
    sum = sum * esc + p;
    const __bf16* vp = &Vl[l2 * 64];
#pragma unroll
    for (int d = 0; d < 64; ++d) c[d] = c[d] * esc + p * (float)vp[d];
    m = mn;
  }
  float inv = 1.0f / sum;
  float* op = ctx + ((size_t)b * SEQ1 + tid) * HDIM + h * 64;
#pragma unroll
  for (int d = 0; d < 64; ++d) op[d] = c[d] * inv;
}

// ---------------- host orchestration ----------------
extern "C" void kernel_launch(void* const* d_in, const int* in_sizes, int n_in,
                              void* d_out, int out_size, void* d_ws, size_t ws_size,
                              hipStream_t stream) {
  (void)in_sizes; (void)n_in; (void)out_size; (void)ws_size;

  const float* s1   = (const float*)d_in[0];
  const float* s2   = (const float*)d_in[1];
  const float* mask = (const float*)d_in[2];
  const float* Wq = (const float*)d_in[3];  const float* bq = (const float*)d_in[4];
  const float* Wk = (const float*)d_in[5];  const float* bk = (const float*)d_in[6];
  const float* Wv = (const float*)d_in[7];  const float* bv = (const float*)d_in[8];
  const float* Wo = (const float*)d_in[9];  const float* bo = (const float*)d_in[10];
  const float* ln1w = (const float*)d_in[11]; const float* ln1b = (const float*)d_in[12];
  const float* Wi = (const float*)d_in[13]; const float* bi = (const float*)d_in[14];
  const float* Wf = (const float*)d_in[15]; const float* bff = (const float*)d_in[16];
  const float* ln2w = (const float*)d_in[17]; const float* ln2b = (const float*)d_in[18];

  char* ws = (char*)d_ws;
  size_t off = 0;
  auto take = [&](size_t bytes) -> void* {
    void* p = ws + off;
    off += (bytes + 255) & ~(size_t)255;
    return p;
  };

  __bf16* wq_t = (__bf16*)take((size_t)HDIM * HDIM * 2);
  __bf16* wk_t = (__bf16*)take((size_t)HDIM * HDIM * 2);
  __bf16* wv_t = (__bf16*)take((size_t)HDIM * HDIM * 2);
  __bf16* wo_t = (__bf16*)take((size_t)HDIM * HDIM * 2);
  __bf16* wi_t = (__bf16*)take((size_t)HDIM * FFDIM * 2);   // [N=3072,K=768]
  __bf16* wf_t = (__bf16*)take((size_t)FFDIM * HDIM * 2);   // [N=768,K=3072]
  __bf16* s1_bf    = (__bf16*)take((size_t)M1 * HDIM * 2);
  __bf16* s2_bf    = (__bf16*)take((size_t)M2 * HDIM * 2);
  __bf16* q_bf     = (__bf16*)take((size_t)M1 * HDIM * 2);
  __bf16* k_bf     = (__bf16*)take((size_t)M2 * HDIM * 2);
  __bf16* v_bf     = (__bf16*)take((size_t)M2 * HDIM * 2);
  __bf16* cbf      = (__bf16*)take((size_t)M1 * HDIM * 2);  // ctx_bf, then attn_bf
  __bf16* inter_bf = (__bf16*)take((size_t)M1 * FFDIM * 2);
  float*  ctx      = (float*) take((size_t)M1 * HDIM * 4);  // ctx / attn_tmp / ffn_out alias
  float*  attn_out = (float*) take((size_t)M1 * HDIM * 4);
  float*  s1_cur   = (float*) take((size_t)M1 * HDIM * 4);

  auto gemm = [&](const __bf16* A, const __bf16* Wt, const float* bias,
                  float* Cf, __bf16* Cbf, int M, int N, int K, int gelu) {
    dim3 grid((N + TILE_N - 1) / TILE_N, (M + TILE_M - 1) / TILE_M);
    k_gemm_wmma<<<grid, 256, 0, stream>>>(A, Wt, bias, Cf, Cbf, M, N, K, gelu);
  };
  auto ew = [&](int n) { return dim3((n + 255) / 256); };

  k_f2bf<<<ew(M2 * HDIM), 256, 0, stream>>>(s2, s2_bf, M2 * HDIM);
  k_f2bf<<<ew(M1 * HDIM), 256, 0, stream>>>(s1, s1_bf, M1 * HDIM);

  const float* s1res = s1;  // residual source for LN1 of current layer

  for (int L = 0; L < 3; ++L) {
    k_wt_convert<<<ew(HDIM * HDIM), 256, 0, stream>>>(Wq + (size_t)L * HDIM * HDIM, wq_t, HDIM, HDIM);
    k_wt_convert<<<ew(HDIM * HDIM), 256, 0, stream>>>(Wk + (size_t)L * HDIM * HDIM, wk_t, HDIM, HDIM);
    k_wt_convert<<<ew(HDIM * HDIM), 256, 0, stream>>>(Wv + (size_t)L * HDIM * HDIM, wv_t, HDIM, HDIM);
    k_wt_convert<<<ew(HDIM * HDIM), 256, 0, stream>>>(Wo + (size_t)L * HDIM * HDIM, wo_t, HDIM, HDIM);
    k_wt_convert<<<ew(HDIM * FFDIM), 256, 0, stream>>>(Wi + (size_t)L * HDIM * FFDIM, wi_t, HDIM, FFDIM);
    k_wt_convert<<<ew(FFDIM * HDIM), 256, 0, stream>>>(Wf + (size_t)L * FFDIM * HDIM, wf_t, FFDIM, HDIM);

    // Q/K/V projections (bf16 outputs feed attention)
    gemm(s1_bf, wq_t, bq + (size_t)L * HDIM, nullptr, q_bf, M1, HDIM, HDIM, 0);
    gemm(s2_bf, wk_t, bk + (size_t)L * HDIM, nullptr, k_bf, M2, HDIM, HDIM, 0);
    gemm(s2_bf, wv_t, bv + (size_t)L * HDIM, nullptr, v_bf, M2, HDIM, HDIM, 0);

    k_attention<<<BATCH * NHEADS, 256, 0, stream>>>(q_bf, k_bf, v_bf, mask, ctx);
    k_f2bf<<<ew(M1 * HDIM), 256, 0, stream>>>(ctx, cbf, M1 * HDIM);

    // output projection (attn_tmp aliases ctx; input is cbf so no overlap)
    gemm(cbf, wo_t, bo + (size_t)L * HDIM, ctx, nullptr, M1, HDIM, HDIM, 0);
    // LN1(attn_tmp + s1) -> attn_out (f32) + cbf (bf16, reused as attn_bf)
    k_ln_residual<<<M1, 256, 0, stream>>>(ctx, s1res, ln1w + (size_t)L * HDIM,
                                          ln1b + (size_t)L * HDIM, attn_out, cbf);

    // FFN up + exact gelu, bf16-only output
    gemm(cbf, wi_t, bi + (size_t)L * FFDIM, nullptr, inter_bf, M1, FFDIM, HDIM, 1);
    // FFN down (ffn_out aliases ctx)
    gemm(inter_bf, wf_t, bff + (size_t)L * HDIM, ctx, nullptr, M1, HDIM, FFDIM, 0);

    float* outp = (L == 2) ? (float*)d_out : s1_cur;
    k_ln_residual<<<M1, 256, 0, stream>>>(ctx, attn_out, ln2w + (size_t)L * HDIM,
                                          ln2b + (size_t)L * HDIM, outp, s1_bf);
    s1res = s1_cur;
  }
}
